// UV_Aggregator_35210141892983
// MI455X (gfx1250) — compile-verified, matrix operand order checked by reading, not run on hardware
//
#include <hip/hip_runtime.h>
#include <hip/hip_bf16.h>

// ---------------------------------------------------------------------------
// Types for WMMA operands
// ---------------------------------------------------------------------------
typedef __attribute__((ext_vector_type(16))) _Float16 v16h;
typedef __attribute__((ext_vector_type(8)))  _Float16 half8;
typedef __attribute__((ext_vector_type(8)))  float    v8f;

#define ENC_NEG_INF 0x007FFFFFu   // order-preserving encoding of -inf

// padded LDS strides (bank-conflict avoidance; rows stay 16B aligned)
#define IN_STRIDE  136   // 32x128 tile rows  -> 272B/row (4-bank rotation)
#define H_STRIDE   72    // 32x64 tile rows   -> 144B/row (36-bank rotation)
#define RED_STRIDE 17

__device__ __forceinline__ unsigned enc_f32(float f) {
    unsigned b = __float_as_uint(f);
    return (b & 0x80000000u) ? ~b : (b | 0x80000000u);
}
__device__ __forceinline__ float dec_f32(unsigned u) {
    unsigned b = (u & 0x80000000u) ? (u & 0x7FFFFFFFu) : ~u;
    return __uint_as_float(b);
}

// ---------------------------------------------------------------------------
// WMMA fragment loader (16-bit A 16x32 striping, ISA 7.12.2):
//   lanes 0-15 : row = lane,    elems 0..7 -> K=k0+0..7,  8..15 -> K=k0+16..23
//   lanes 16-31: row = lane-16, elems 0..7 -> K=k0+8..15, 8..15 -> K=k0+24..31
// B supplied as B^T (weights row-major [N][K]), same striping with N as row.
// ---------------------------------------------------------------------------
__device__ __forceinline__ v16h load_frag(const _Float16* base, int stride, int k0, int lane) {
    int row = lane & 15;
    int hf  = lane >> 4;
    const half8* p0 = (const half8*)(base + row * stride + k0 + hf * 8);
    const half8* p1 = (const half8*)(base + row * stride + k0 + 16 + hf * 8);
    half8 lo = *p0;
    half8 hi = *p1;
    v16h r;
#pragma unroll
    for (int i = 0; i < 8; ++i) { r[i] = lo[i]; r[i + 8] = hi[i]; }
    return r;
}

// ---------------------------------------------------------------------------
// 32xK * Kx64 tile-GEMM (K = 32*KT), two M-tiles sharing each B fragment.
// k-outer loop: acc[nt][mt] persistent, A per-(k,mt), B per-(k,nt) feeds 2 WMMAs.
// out[nt][mt][v] = value(row M = mt*16 + v + 8*(lane>>4), col N = nt*16 + (lane&15))
// ---------------------------------------------------------------------------
template <int KT>
__device__ __forceinline__ void gemm2_relu(const _Float16* At, int astride,
                                           const _Float16* W, int kdim,
                                           const float* bias, int lane,
                                           float out[4][2][8]) {
    v8f acc[4][2];
#pragma unroll
    for (int nt = 0; nt < 4; ++nt)
#pragma unroll
        for (int mt = 0; mt < 2; ++mt)
            acc[nt][mt] = (v8f){0.f, 0.f, 0.f, 0.f, 0.f, 0.f, 0.f, 0.f};

#pragma unroll
    for (int k = 0; k < KT; ++k) {
        v16h a0 = load_frag(At,                astride, k * 32, lane);
        v16h a1 = load_frag(At + 16 * astride, astride, k * 32, lane);
#pragma unroll
        for (int nt = 0; nt < 4; ++nt) {
            v16h b = load_frag(W + (nt * 16) * kdim, kdim, k * 32, lane);
            acc[nt][0] = __builtin_amdgcn_wmma_f32_16x16x32_f16(
                false, a0, false, b, (short)0, acc[nt][0], false, false);
            acc[nt][1] = __builtin_amdgcn_wmma_f32_16x16x32_f16(
                false, a1, false, b, (short)0, acc[nt][1], false, false);
        }
    }
    int col = lane & 15;
#pragma unroll
    for (int nt = 0; nt < 4; ++nt) {
        float bv = bias[nt * 16 + col];
#pragma unroll
        for (int mt = 0; mt < 2; ++mt)
#pragma unroll
            for (int v = 0; v < 8; ++v)
                out[nt][mt][v] = fmaxf(acc[nt][mt][v] + bv, 0.0f);
    }
}

// scatter C-layout results into an f16 LDS tile
__device__ __forceinline__ void scatter_tile(_Float16* dst, int dstride, int lane,
                                             const float out[4][2][8]) {
    int col = lane & 15;
    int hf  = lane >> 4;
#pragma unroll
    for (int nt = 0; nt < 4; ++nt)
#pragma unroll
        for (int mt = 0; mt < 2; ++mt)
#pragma unroll
            for (int v = 0; v < 8; ++v)
                dst[(mt * 16 + v + 8 * hf) * dstride + nt * 16 + col] =
                    (_Float16)out[nt][mt][v];
}

// ---------------------------------------------------------------------------
// Kernel 0: weights fp32 -> f16 (w1 64x128 | w2 64x64 | aw1 64x128 | aw2 64x64)
// ---------------------------------------------------------------------------
__global__ void prep_weights_kernel(const float* w1, const float* w2,
                                    const float* aw1, const float* aw2,
                                    _Float16* w1h, _Float16* w2h,
                                    _Float16* aw1h, _Float16* aw2h) {
    int i = blockIdx.x * blockDim.x + threadIdx.x;
    if (i < 8192)        w1h[i]          = (_Float16)w1[i];
    else if (i < 12288)  w2h[i - 8192]   = (_Float16)w2[i - 8192];
    else if (i < 20480)  aw1h[i - 12288] = (_Float16)aw1[i - 12288];
    else if (i < 24576)  aw2h[i - 20480] = (_Float16)aw2[i - 20480];
}

// Kernel 1: init segment buffers + n_feat accumulator
__global__ void init_kernel(unsigned* seg_max_u, float* seg_sum, float* n_feat, int N) {
    long i = (long)blockIdx.x * blockDim.x + threadIdx.x;
    long tot = (long)N * 64;
    if (i < tot) n_feat[i] = 0.0f;
    if (i < N) { seg_sum[i] = 0.0f; seg_max_u[i] = ENC_NEG_INF; }
}

// ---------------------------------------------------------------------------
// Kernel 2: fused per-edge pipeline (WMMA). One wave handles 32 edges.
// ---------------------------------------------------------------------------
#define NW 4   // waves per block (blockDim.x = 128)

__global__ void __launch_bounds__(128)
edge_kernel(const int* __restrict__ row_idxs, const int* __restrict__ col_idxs,
            const int* __restrict__ ratings,
            const float* __restrict__ v2e, const float* __restrict__ u2e,
            const float* __restrict__ r2e,
            const _Float16* __restrict__ w1h, const float* __restrict__ b1,
            const _Float16* __restrict__ w2h, const float* __restrict__ b2,
            const _Float16* __restrict__ aw1h, const float* __restrict__ ab1,
            const _Float16* __restrict__ aw2h, const float* __restrict__ ab2,
            const float* __restrict__ aw3, const float* __restrict__ ab3,
            _Float16* __restrict__ oh_h, float* __restrict__ logits,
            unsigned* __restrict__ seg_max_u, int E) {
    __shared__ alignas(16) _Float16 in_tile[NW][32 * IN_STRIDE];  // [edge][2d] f16
    __shared__ alignas(16) _Float16 h_tile[NW][32 * H_STRIDE];    // hidden f16
    __shared__ float red[NW][32 * RED_STRIDE];                    // logit reduction

    const int w    = threadIdx.x >> 5;
    const int lane = threadIdx.x & 31;
    const long e0  = ((long)blockIdx.x * NW + w) * 32;
    if (e0 >= E) return;

    const int col = lane & 15;
    const int hf  = lane >> 4;

    // per-lane edge bookkeeping (lane L owns edge e0+L for staging / logits)
    long e_own = e0 + lane; if (e_own >= E) e_own = E - 1;
    const int c_own = col_idxs[e_own];

    // ---- stage [emb_uv | emb_r]: lane L stages edge L's full 128-col row ----
    {
        const float4* su = (const float4*)(v2e + (long)row_idxs[e_own] * 64);
        const float4* sr = (const float4*)(r2e + (long)ratings[e_own] * 64);
        half8* dst = (half8*)&in_tile[w][lane * IN_STRIDE];
#pragma unroll
        for (int j = 0; j < 8; ++j) {
            float4 fa = su[2 * j], fb = su[2 * j + 1];
            dst[j] = (half8){(_Float16)fa.x, (_Float16)fa.y, (_Float16)fa.z, (_Float16)fa.w,
                             (_Float16)fb.x, (_Float16)fb.y, (_Float16)fb.z, (_Float16)fb.w};
        }
#pragma unroll
        for (int j = 0; j < 8; ++j) {
            float4 fa = sr[2 * j], fb = sr[2 * j + 1];
            dst[8 + j] = (half8){(_Float16)fa.x, (_Float16)fa.y, (_Float16)fa.z, (_Float16)fa.w,
                                 (_Float16)fb.x, (_Float16)fb.y, (_Float16)fb.z, (_Float16)fb.w};
        }
    }
    asm volatile("s_wait_dscnt 0" ::: "memory");

    // ---- edge MLP layer 1: [32x128] x [128x64] ----
    float o1[4][2][8];
    gemm2_relu<4>(&in_tile[w][0], IN_STRIDE, w1h, 128, b1, lane, o1);
    scatter_tile(&h_tile[w][0], H_STRIDE, lane, o1);

    // ---- stage rep = u2e[col_idxs] into cols 64..127 (overwrites emb_r) ----
    {
        const float4* su = (const float4*)(u2e + (long)c_own * 64);
        half8* dst = (half8*)&in_tile[w][lane * IN_STRIDE];
#pragma unroll
        for (int j = 0; j < 8; ++j) {
            float4 fa = su[2 * j], fb = su[2 * j + 1];
            dst[8 + j] = (half8){(_Float16)fa.x, (_Float16)fa.y, (_Float16)fa.z, (_Float16)fa.w,
                                 (_Float16)fb.x, (_Float16)fb.y, (_Float16)fb.z, (_Float16)fb.w};
        }
    }
    asm volatile("s_wait_dscnt 0" ::: "memory");

    // ---- edge MLP layer 2: [32x64] x [64x64] -> ohistory ----
    float oh[4][2][8];
    gemm2_relu<2>(&h_tile[w][0], H_STRIDE, w2h, 64, b2, lane, oh);
#pragma unroll
    for (int nt = 0; nt < 4; ++nt)
#pragma unroll
        for (int mt = 0; mt < 2; ++mt)
#pragma unroll
            for (int v = 0; v < 8; ++v) {
                int erow = mt * 16 + v + 8 * hf;
                _Float16 hv = (_Float16)oh[nt][mt][v];
                in_tile[w][erow * IN_STRIDE + nt * 16 + col] = hv; // attn input 0..63
                long e = e0 + erow;
                if (e < E) oh_h[e * 64 + nt * 16 + col] = hv;      // persist
            }
    asm volatile("s_wait_dscnt 0" ::: "memory");

    // ---- attention layer 1: [32x128] x [128x64] ----
    float a1[4][2][8];
    gemm2_relu<4>(&in_tile[w][0], IN_STRIDE, aw1h, 128, ab1, lane, a1);
    scatter_tile(&h_tile[w][0], H_STRIDE, lane, a1);
    asm volatile("s_wait_dscnt 0" ::: "memory");

    // ---- attention layer 2: [32x64] x [64x64] ----
    float a2[4][2][8];
    gemm2_relu<2>(&h_tile[w][0], H_STRIDE, aw2h, 64, ab2, lane, a2);

    // ---- logits: a2 . att_w3 (64-wide dot per edge) ----
    float s[2][8];
#pragma unroll
    for (int mt = 0; mt < 2; ++mt)
#pragma unroll
        for (int v = 0; v < 8; ++v) s[mt][v] = 0.0f;
#pragma unroll
    for (int nt = 0; nt < 4; ++nt) {
        float w3 = aw3[nt * 16 + col];
#pragma unroll
        for (int mt = 0; mt < 2; ++mt)
#pragma unroll
            for (int v = 0; v < 8; ++v) s[mt][v] += a2[nt][mt][v] * w3;
    }
#pragma unroll
    for (int mt = 0; mt < 2; ++mt)
#pragma unroll
        for (int v = 0; v < 8; ++v)
            red[w][(mt * 16 + v + 8 * hf) * RED_STRIDE + col] = s[mt][v];
    asm volatile("s_wait_dscnt 0" ::: "memory");

    // one edge per lane (all 32 lanes active)
    if (e0 + lane < E) {
        float lg = ab3[0];
#pragma unroll
        for (int j = 0; j < 16; ++j) lg += red[w][lane * RED_STRIDE + j];
        logits[e0 + lane] = lg;
        atomicMax(&seg_max_u[c_own], enc_f32(lg));
    }
}

// ---------------------------------------------------------------------------
// Kernel 3: e = exp(logit - segmax); seg_sum += e
// ---------------------------------------------------------------------------
__global__ void softmax_sum_kernel(const float* __restrict__ logits,
                                   const int* __restrict__ col_idxs,
                                   const unsigned* __restrict__ seg_max_u,
                                   float* __restrict__ evals,
                                   float* __restrict__ seg_sum, int E) {
    long e = (long)blockIdx.x * blockDim.x + threadIdx.x;
    if (e >= E) return;
    int  c  = col_idxs[e];
    float m = dec_f32(seg_max_u[c]);
    float ev = __expf(logits[e] - m);
    evals[e] = ev;
    atomicAdd(&seg_sum[c], ev);
}

// ---------------------------------------------------------------------------
// Kernel 4: msg = ohistory * att; scatter-sum. One wave per edge, 2 feats/lane.
// ---------------------------------------------------------------------------
__global__ void aggregate_kernel(const _Float16* __restrict__ oh_h,
                                 const float* __restrict__ evals,
                                 const float* __restrict__ seg_sum,
                                 const int* __restrict__ col_idxs,
                                 float* __restrict__ n_feat, int E) {
    long gid  = (long)blockIdx.x * blockDim.x + threadIdx.x;
    long edge = gid >> 5;
    if (edge >= E) return;
    int lane = (int)(gid & 31);
    int c    = col_idxs[edge];
    float att = evals[edge] / seg_sum[c];
    const _Float16* op = oh_h + edge * 64 + lane * 2;
    float m0 = (float)op[0] * att;
    float m1 = (float)op[1] * att;
    float* dst = n_feat + (long)c * 64 + lane * 2;
    atomicAdd(dst + 0, m0);
    atomicAdd(dst + 1, m1);
}

// Kernel 5: out = n_feat[nodes]
__global__ void gather_out_kernel(const float* __restrict__ n_feat,
                                  const int* __restrict__ nodes,
                                  float* __restrict__ out, int B) {
    long gid = (long)blockIdx.x * blockDim.x + threadIdx.x;
    if (gid >= (long)B * 64) return;
    long i = gid >> 6;
    int  f = (int)(gid & 63);
    out[gid] = n_feat[(long)nodes[i] * 64 + f];
}

// ---------------------------------------------------------------------------
extern "C" void kernel_launch(void* const* d_in, const int* in_sizes, int n_in,
                              void* d_out, int out_size, void* d_ws, size_t ws_size,
                              hipStream_t stream) {
    const int*   nodes    = (const int*)d_in[0];
    const int*   row_idxs = (const int*)d_in[1];
    const int*   col_idxs = (const int*)d_in[2];
    const int*   ratings  = (const int*)d_in[3];
    const float* v2e = (const float*)d_in[4];
    const float* u2e = (const float*)d_in[5];
    const float* r2e = (const float*)d_in[6];
    const float* w1  = (const float*)d_in[7];
    const float* b1  = (const float*)d_in[8];
    const float* w2  = (const float*)d_in[9];
    const float* b2  = (const float*)d_in[10];
    const float* aw1 = (const float*)d_in[11];
    const float* ab1 = (const float*)d_in[12];
    const float* aw2 = (const float*)d_in[13];
    const float* ab2 = (const float*)d_in[14];
    const float* aw3 = (const float*)d_in[15];
    const float* ab3 = (const float*)d_in[16];

    const int E = in_sizes[1];
    const int N = in_sizes[5] / 64;   // NUM_NODES from u2e_w
    const int B = in_sizes[0];

    // workspace carve-out (256B aligned slices)
    char*  ws  = (char*)d_ws;
    size_t off = 0;
    auto take = [&](size_t bytes) -> char* {
        char* p = ws + off;
        off = (off + bytes + 255) & ~(size_t)255;
        return p;
    };
    _Float16* oh_h      = (_Float16*)take((size_t)E * 64 * sizeof(_Float16));
    float*    logits    = (float*)take((size_t)E * sizeof(float));
    float*    evals     = (float*)take((size_t)E * sizeof(float));
    unsigned* seg_max_u = (unsigned*)take((size_t)N * sizeof(unsigned));
    float*    seg_sum   = (float*)take((size_t)N * sizeof(float));
    float*    n_feat    = (float*)take((size_t)N * 64 * sizeof(float));
    _Float16* w1h       = (_Float16*)take(8192 * sizeof(_Float16));
    _Float16* w2h       = (_Float16*)take(4096 * sizeof(_Float16));
    _Float16* aw1h      = (_Float16*)take(8192 * sizeof(_Float16));
    _Float16* aw2h      = (_Float16*)take(4096 * sizeof(_Float16));
    (void)ws_size; (void)n_in; (void)out_size;

    // 0) weight conversion
    prep_weights_kernel<<<(24576 + 255) / 256, 256, 0, stream>>>(
        w1, w2, aw1, aw2, w1h, w2h, aw1h, aw2h);

    // 1) init accumulators
    long itot = (long)N * 64;
    init_kernel<<<(unsigned)((itot + 255) / 256), 256, 0, stream>>>(
        seg_max_u, seg_sum, n_feat, N);

    // 2) fused edge pipeline (WMMA), 32 edges/wave, 4 waves/block
    int edge_blocks = (E + 32 * NW - 1) / (32 * NW);
    edge_kernel<<<edge_blocks, 128, 0, stream>>>(
        row_idxs, col_idxs, ratings, v2e, u2e, r2e,
        w1h, b1, w2h, b2, aw1h, ab1, aw2h, ab2, aw3, ab3,
        oh_h, logits, seg_max_u, E);

    // 3) segment softmax denominator
    softmax_sum_kernel<<<(E + 255) / 256, 256, 0, stream>>>(
        logits, col_idxs, seg_max_u, evals, seg_sum, E);

    // 4) weighted scatter-sum
    long atot = (long)E * 32;
    aggregate_kernel<<<(unsigned)((atot + 255) / 256), 256, 0, stream>>>(
        oh_h, evals, seg_sum, col_idxs, n_feat, E);

    // 5) gather output rows
    gather_out_kernel<<<(B * 64 + 255) / 256, 256, 0, stream>>>(
        n_feat, nodes, (float*)d_out, B);
}
